// RelationModule_60060822667746
// MI455X (gfx1250) — matrix operand order repaired and checked by reading, not compiled, not run on hardware
//
#include <hip/hip_runtime.h>

#define NPROP 1024
#define DAPP  1024
#define DK    64
#define DG    64
#define NRELS 16
#define NNELE (NPROP * NPROP)

typedef __attribute__((ext_vector_type(16))) _Float16 v16h;
typedef __attribute__((ext_vector_type(8)))  _Float16 v8h;
typedef __attribute__((ext_vector_type(8)))  float    v8f;
typedef __attribute__((ext_vector_type(4)))  float    v4f;

// ---------- WMMA helpers (16x16x32 f16 -> f32) ----------
static __device__ __forceinline__ v8f wmma32(v16h a, v16h b, v8f c) {
  return __builtin_amdgcn_wmma_f32_16x16x32_f16(
      /*neg_a=*/false, a, /*neg_b=*/false, b,
      /*c_mod=*/(short)0, c, /*reuse_a=*/false, /*reuse_b=*/false);
}

// A-matrix 16x32 f16 fragment: lane holds row = lane%16,
// halves 0..15 = K[2b..2b+16) contiguous, b = (lane>=16)*8
static __device__ __forceinline__ v16h frag_a_h(const _Float16* rowk, int b) {
  v8h lo = *(const v8h*)(rowk + 2 * b);
  v8h hi = *(const v8h*)(rowk + 2 * b + 8);
  return __builtin_shufflevector(lo, hi, 0, 1, 2, 3, 4, 5, 6, 7,
                                 8, 9, 10, 11, 12, 13, 14, 15);
}

// B-matrix 32x16 f16 fragment (from Y stored as [n][k], n = output column):
// lane holds column n = lane%16, 16 contiguous K starting at boff = (lane>=16)*16
static __device__ __forceinline__ v16h frag_b_h(const _Float16* rowk, int boff) {
  v8h lo = *(const v8h*)(rowk + boff);
  v8h hi = *(const v8h*)(rowk + boff + 8);
  return __builtin_shufflevector(lo, hi, 0, 1, 2, 3, 4, 5, 6, 7,
                                 8, 9, 10, 11, 12, 13, 14, 15);
}

// 8 f32 -> v8h, nontemporal (streamed pe)
static __device__ __forceinline__ v8h cvt8_nt(const float* p) {
  v4f x = __builtin_nontemporal_load((const v4f*)p);
  v4f y = __builtin_nontemporal_load((const v4f*)(p + 4));
  v8h r;
#pragma unroll
  for (int i = 0; i < 4; ++i) { r[i] = (_Float16)x[i]; r[4 + i] = (_Float16)y[i]; }
  return r;
}
static __device__ __forceinline__ v8h cvt8(const float* p) {
  v8h r;
#pragma unroll
  for (int i = 0; i < 8; ++i) r[i] = (_Float16)p[i];
  return r;
}
static __device__ __forceinline__ v16h frag_a_f32nt(const float* rowk, int b) {
  v8h lo = cvt8_nt(rowk + 2 * b);
  v8h hi = cvt8_nt(rowk + 2 * b + 8);
  return __builtin_shufflevector(lo, hi, 0, 1, 2, 3, 4, 5, 6, 7,
                                 8, 9, 10, 11, 12, 13, 14, 15);
}
static __device__ __forceinline__ v16h frag_b_f32(const float* rowk, int boff) {
  v8h lo = cvt8(rowk + boff);
  v8h hi = cvt8(rowk + boff + 8);
  return __builtin_shufflevector(lo, hi, 0, 1, 2, 3, 4, 5, 6, 7,
                                 8, 9, 10, 11, 12, 13, 14, 15);
}

// ---------- Kernel 0: f32 -> f16 conversion ----------
__global__ void cvt_kernel(const float* __restrict__ src,
                           _Float16* __restrict__ dst, int n) {
  int i = blockIdx.x * blockDim.x + threadIdx.x;
  if (i < n) dst[i] = (_Float16)src[i];
}

// ---------- Kernel G: log-gate for all 16 relations via WMMA ----------
// lg[r][n][m] = log(max(pe[n,m,:].WG_w[r,:] + WG_b[r], 1e-6))  (f16)
// One wave handles 16 consecutive (n,m) pairs x all 16 relations with 2 WMMAs.
__global__ void gate_kernel(const float* __restrict__ pe,
                            const float* __restrict__ WGw,
                            const float* __restrict__ WGb,
                            _Float16* __restrict__ lg) {
  const int lane = threadIdx.x & 31;
  const int wave = threadIdx.x >> 5;
  const long waveId = (long)blockIdx.x * (blockDim.x >> 5) + wave;
  const long nmBase = waveId * 16;
  const int l15 = lane & 15;   // A row (nm-local) source / B row (relation) source
  const int hiw = lane >> 4;
  const int ab = hiw * 8, bb = hiw * 16;

  const float* perow = pe + (size_t)(nmBase + l15) * DG;
  const float* wrow = WGw + (size_t)l15 * DG;

  v8f acc = {};
  acc = wmma32(frag_a_f32nt(perow + 0, ab), frag_b_f32(wrow + 0, bb), acc);
  acc = wmma32(frag_a_f32nt(perow + 32, ab), frag_b_f32(wrow + 32, bb), acc);

  const float bias = WGb[l15];  // lane's D column = relation l15
  v8h pk;
#pragma unroll
  for (int v = 0; v < 8; ++v) {
    float x = acc[v] + bias;
    x = fmaxf(x, 1e-6f);               // log(max(relu(x),1e-6)) == log(max(x,1e-6))
    pk[v] = (_Float16)__logf(x);
  }
  // D rows v..v+7 (+8 for upper half-wave) are consecutive nm -> one 16B store
  _Float16* dstp = lg + (size_t)l15 * NNELE + nmBase + hiw * 8;
  __builtin_nontemporal_store(pk, (v8h*)dstp);
}

// ---------- Kernel 1: per-relation K/Q/V projections ----------
// D[n][d] = f_a[n,:] . W[r,d,:] + b[r,d] ; stored row-major [r][n][d]
// or transposed [r][d][n] (for V, so it feeds the AV B-operand).
__global__ void proj_kernel(const _Float16* __restrict__ fa_h,
                            const _Float16* __restrict__ Wh,  // [NR][DK][DAPP]
                            const float* __restrict__ Wb,     // [NR][DK]
                            _Float16* __restrict__ dst, int transposed) {
  const int lane = threadIdx.x & 31;
  const int wave = threadIdx.x >> 5;  // 0..3 -> d-tile
  const int n0 = blockIdx.x * 16;
  const int r = blockIdx.y;
  const int d0 = wave * 16;
  const int l15 = lane & 15, hiw = lane >> 4;
  const int ab = hiw * 8, bb = hiw * 16, off = hiw * 8;

  const _Float16* arow = fa_h + (size_t)(n0 + l15) * DAPP;
  const _Float16* brow = Wh + ((size_t)r * DK + d0 + l15) * DAPP;

  v8f acc = {};
#pragma unroll 4
  for (int k0 = 0; k0 < DAPP; k0 += 32)
    acc = wmma32(frag_a_h(arow + k0, ab), frag_b_h(brow + k0, bb), acc);

  const float bias = Wb[r * DK + d0 + l15];
  if (!transposed) {
#pragma unroll
    for (int v = 0; v < 8; ++v)
      dst[((size_t)r * NPROP + n0 + off + v) * DK + d0 + l15] =
          (_Float16)(acc[v] + bias);
  } else {
    v8h pk;
#pragma unroll
    for (int v = 0; v < 8; ++v) pk[v] = (_Float16)(acc[v] + bias);
    *(v8h*)(dst + ((size_t)r * DK + d0 + l15) * NPROP + n0 + off) = pk;
  }
}

// ---------- Kernel 2: fused scores + log-gate + softmax + AV + residual ----------
// Block = 128 threads (4 waves), handles (relation r, 16-row tile n0).
__global__ void attn_kernel(const _Float16* __restrict__ Kh,  // [NR][N][DK]
                            const _Float16* __restrict__ Qh,  // [NR][N][DK]
                            const _Float16* __restrict__ Vt,  // [NR][DK][N]
                            const _Float16* __restrict__ lg,  // [NR][N][N]
                            const float* __restrict__ fa,     // [N][DAPP]
                            float* __restrict__ out) {        // [N][DAPP]
  __shared__ __attribute__((aligned(16))) _Float16 wbuf[16 * NPROP];  // 32 KB
  __shared__ float red[128];
  __shared__ float rinv[16];

  const int tid = threadIdx.x;
  const int lane = tid & 31, wave = tid >> 5;
  const int n0 = blockIdx.x * 16;
  const int r = blockIdx.y;
  const int l15 = lane & 15, hiw = lane >> 4;
  const int ab = hiw * 8, bb = hiw * 16, off = hiw * 8;

  // K-row A-fragments (shared by all m-tiles)
  const _Float16* krow = Kh + ((size_t)r * NPROP + n0 + l15) * DK;
  const v16h ka0 = frag_a_h(krow + 0, ab);
  const v16h ka1 = frag_a_h(krow + 32, ab);
  const _Float16* lgp = lg + (size_t)r * NNELE;

  // Phase 1: w[n][m] = scores/8 + lg, staged in LDS (each wave: 16 m-tiles)
  for (int mt = wave * 16; mt < wave * 16 + 16; ++mt) {
    const int m0 = mt * 16;
    const _Float16* qrow = Qh + ((size_t)r * NPROP + m0 + l15) * DK;
    v8f acc = {};
    acc = wmma32(ka0, frag_b_h(qrow + 0, bb), acc);
    acc = wmma32(ka1, frag_b_h(qrow + 32, bb), acc);
#pragma unroll
    for (int v = 0; v < 8; ++v) {
      float lgv = (float)__builtin_nontemporal_load(
          lgp + (size_t)(n0 + off + v) * NPROP + m0 + l15);
      wbuf[(off + v) * NPROP + m0 + l15] = (_Float16)(acc[v] * 0.125f + lgv);
    }
  }
  __syncthreads();

  // Phase 2: softmax over m (two-pass; 8 threads per row)
  const int row = tid >> 3, sub = tid & 7;
  _Float16* wrow = wbuf + row * NPROP + sub * 128;
  float pmax = -3.0e38f;
  for (int j = 0; j < 128; j += 8) {
    v8h x = *(const v8h*)(wrow + j);
#pragma unroll
    for (int t = 0; t < 8; ++t) pmax = fmaxf(pmax, (float)x[t]);
  }
  red[tid] = pmax;
  __syncthreads();
  float mrow = red[row * 8];
#pragma unroll
  for (int j = 1; j < 8; ++j) mrow = fmaxf(mrow, red[row * 8 + j]);
  __syncthreads();  // everyone done reading red before reuse
  float psum = 0.f;
  for (int j = 0; j < 128; j += 8) {
    v8h x = *(const v8h*)(wrow + j);
    v8h e;
#pragma unroll
    for (int t = 0; t < 8; ++t) {
      float ev = __expf((float)x[t] - mrow);
      psum += ev;
      e[t] = (_Float16)ev;
    }
    *(v8h*)(wrow + j) = e;
  }
  red[tid] = psum;
  __syncthreads();
  float s = 0.f;
#pragma unroll
  for (int j = 0; j < 8; ++j) s += red[row * 8 + j];
  if (sub == 0) rinv[row] = 1.0f / s;
  __syncthreads();

  // Phase 3: AV (1/sum folded into output scale) + residual. Wave -> d-tile.
  const int d0 = wave * 16;
  const _Float16* vrow = Vt + ((size_t)r * DK + d0 + l15) * NPROP;
  const _Float16* arow = wbuf + l15 * NPROP;
  v8f acc = {};
#pragma unroll 4
  for (int k0 = 0; k0 < NPROP; k0 += 32)
    acc = wmma32(frag_a_h(arow + k0, ab), frag_b_h(vrow + k0, bb), acc);
#pragma unroll
  for (int v = 0; v < 8; ++v) {
    const int n = n0 + off + v;
    const int c = r * DK + d0 + l15;
    out[(size_t)n * DAPP + c] = acc[v] * rinv[off + v] + fa[(size_t)n * DAPP + c];
  }
}

extern "C" void kernel_launch(void* const* d_in, const int* in_sizes, int n_in,
                              void* d_out, int out_size, void* d_ws, size_t ws_size,
                              hipStream_t stream) {
  (void)in_sizes; (void)n_in; (void)out_size; (void)ws_size;
  const float* f_a = (const float*)d_in[0];
  const float* pe  = (const float*)d_in[1];
  const float* WGw = (const float*)d_in[2];
  const float* WGb = (const float*)d_in[3];
  const float* WKw = (const float*)d_in[4];
  const float* WKb = (const float*)d_in[5];
  const float* WQw = (const float*)d_in[6];
  const float* WQb = (const float*)d_in[7];
  const float* WVw = (const float*)d_in[8];
  const float* WVb = (const float*)d_in[9];
  float* out = (float*)d_out;

  // Workspace layout (needs 46 MB)
  char* ws = (char*)d_ws;
  _Float16* fa_h = (_Float16*)(ws + ((size_t)0 << 20));
  _Float16* WKh  = (_Float16*)(ws + ((size_t)2 << 20));
  _Float16* WQh  = (_Float16*)(ws + ((size_t)4 << 20));
  _Float16* WVh  = (_Float16*)(ws + ((size_t)6 << 20));
  _Float16* Kh   = (_Float16*)(ws + ((size_t)8 << 20));
  _Float16* Qh   = (_Float16*)(ws + ((size_t)10 << 20));
  _Float16* Vt   = (_Float16*)(ws + ((size_t)12 << 20));
  _Float16* lg   = (_Float16*)(ws + ((size_t)14 << 20));  // 32 MB

  const int NC = NPROP * DAPP;  // 1M elements each
  cvt_kernel<<<NC / 256, 256, 0, stream>>>(f_a, fa_h, NC);
  cvt_kernel<<<NC / 256, 256, 0, stream>>>(WKw, WKh, NC);
  cvt_kernel<<<NC / 256, 256, 0, stream>>>(WQw, WQh, NC);
  cvt_kernel<<<NC / 256, 256, 0, stream>>>(WVw, WVh, NC);

  // 1,048,576 nm pairs / 16 per wave / 8 waves per block = 8192 blocks
  gate_kernel<<<8192, 256, 0, stream>>>(pe, WGw, WGb, lg);

  proj_kernel<<<dim3(NPROP / 16, NRELS), 128, 0, stream>>>(fa_h, WKh, WKb, Kh, 0);
  proj_kernel<<<dim3(NPROP / 16, NRELS), 128, 0, stream>>>(fa_h, WQh, WQb, Qh, 0);
  proj_kernel<<<dim3(NPROP / 16, NRELS), 128, 0, stream>>>(fa_h, WVh, WVb, Vt, 1);

  attn_kernel<<<dim3(NPROP / 16, NRELS), 128, 0, stream>>>(Kh, Qh, Vt, lg, f_a, out);
}
